// MultiHeadAttentionEfficient_84894323572775
// MI455X (gfx1250) — matrix operand order repaired and checked by reading, not compiled, not run on hardware
//
#include <hip/hip_runtime.h>
#include <hip/hip_bf16.h>
#include <stdint.h>

#define N_EMBD 1024
#define N_HEADS 16
#define HEAD 64
#define BATCH 4
#define SEQ 2048
#define TOKENS (BATCH * SEQ) // 8192

typedef __attribute__((ext_vector_type(16))) _Float16 v16h;
typedef __attribute__((ext_vector_type(8)))  _Float16 v8h;
typedef __attribute__((ext_vector_type(8)))  float    v8f;

__device__ __forceinline__ v8f wmma16(v16h a, v16h b, v8f c) {
    // D(16x16,f32) = A(16x32,f16) x B(32x16,f16) + C
    return __builtin_amdgcn_wmma_f32_16x16x32_f16(false, a, false, b,
                                                  (short)0, c, false, false);
}

// CDNA5 async global->LDS copy, 16 bytes. Tracked on ASYNCcnt.
__device__ __forceinline__ void async_copy_b128(uint32_t lds_off, const void* gptr) {
    asm volatile("global_load_async_to_lds_b128 %0, %1, off"
                 :: "v"(lds_off), "v"(gptr) : "memory");
}
__device__ __forceinline__ void wait_async0() {
    asm volatile("s_wait_asynccnt 0x0" ::: "memory");
}

// A-fragment (16x32) from row-major f16 matrix, leading dim lda.
// Per ISA: lane L -> M = L%16; elems 0..7 -> K = 8*(L/16)+0..7,
//          elems 8..15 -> K = 16 + 8*(L/16)+0..7.
__device__ __forceinline__ v16h load_a(const _Float16* A, int lda, int row0, int k0) {
    int lane = threadIdx.x & 31;
    int m = lane & 15, lh = lane >> 4;
    const _Float16* p = A + (size_t)(row0 + m) * lda + k0 + 8 * lh;
    v8h lo = *(const v8h*)p;
    v8h hi = *(const v8h*)(p + 16);
    v16h r;
#pragma unroll
    for (int i = 0; i < 8; ++i) { r[i] = lo[i]; r[8 + i] = hi[i]; }
    return r;
}

// B-fragment (32x16) from row-major f16 matrix [K x N], leading dim ldb.
// Per ISA: lane L -> N = L%16; elems 0..7 -> K = 8*(L/16)+0..7,
//          elems 8..15 -> K = 16 + 8*(L/16)+0..7.
__device__ __forceinline__ v16h load_b(const _Float16* B, int ldb, int k0, int n0) {
    int lane = threadIdx.x & 31;
    v16h r;
#pragma unroll
    for (int i = 0; i < 16; ++i) {
        int k = (i & 7) + 8 * (lane >> 4) + ((i >> 3) << 4);
        r[i] = B[(size_t)(k0 + k) * ldb + n0 + (lane & 15)];
    }
    return r;
}

__global__ void cvt_f32_f16(const float* __restrict__ in, _Float16* __restrict__ out, int n) {
    int i = blockIdx.x * blockDim.x + threadIdx.x;
    int stride = gridDim.x * blockDim.x;
    for (; i < n; i += stride) out[i] = (_Float16)in[i];
}

// ---------------------------------------------------------------------------
// GEMM building block: block = 4 waves, each wave computes a 32x64 tile.
// B tile (32 x 64 f16 = 4KB) shared by all waves: double-buffered in LDS,
// filled with async global->LDS copies overlapped with WMMA on the other buf.
// Double buffering is done with uniform pointer swaps (not kb&1 indexing) and
// the K-loop is kept un-unrolled so accumulators stay pinned to WMMA D-ranges.
// ---------------------------------------------------------------------------

// qkv = x @ W_qkv + b_qkv, scattered into Q[b,h,t,d], Kt[b,h,d,t], V[b,h,t,d]
__global__ __launch_bounds__(128)
void qkv_gemm(const _Float16* __restrict__ X, const _Float16* __restrict__ W,
              const float* __restrict__ bias,
              _Float16* __restrict__ Q, _Float16* __restrict__ Kt, _Float16* __restrict__ V) {
    __shared__ __align__(128) _Float16 sB[2][32 * 64];
    const int ldb = 3 * N_EMBD;
    const int NK = N_EMBD / 32;
    int tid  = threadIdx.x;
    int wave = tid >> 5;
    int lane = tid & 31;
    int row0 = blockIdx.y * 128 + wave * 32;
    int n0   = blockIdx.x * 64;

    // this thread's slice of the B tile: 32 contiguous bytes
    int byt  = tid * 32;
    int brow = byt >> 7;            // 0..31  (k offset within tile)
    int bcol = (byt & 127) >> 1;    // f16 column offset 0..63

    uint32_t lds_cur = (uint32_t)(size_t)&sB[0][0] + byt;
    uint32_t lds_nxt = (uint32_t)(size_t)&sB[1][0] + byt;
    const _Float16* bb_cur = &sB[0][0] + (lane & 15);
    const _Float16* bb_nxt = &sB[1][0] + (lane & 15);
    const _Float16* gsrc = W + (size_t)brow * ldb + n0 + bcol;

    // prologue: stage k-block 0 into buffer 0
    async_copy_b128(lds_cur, gsrc);
    async_copy_b128(lds_cur + 16, gsrc + 8);
    gsrc += (size_t)32 * ldb;

    v8f acc[2][4] = {};
#pragma clang loop unroll(disable)
    for (int kb = 0; kb < NK; ++kb) {
        // global A loads issued before the async wait so they overlap it
        v16h a0 = load_a(X, N_EMBD, row0,      kb * 32);
        v16h a1 = load_a(X, N_EMBD, row0 + 16, kb * 32);
        wait_async0();
        __syncthreads();   // current buffer ready; other buffer fully consumed
        if (kb + 1 < NK) {
            async_copy_b128(lds_nxt, gsrc);
            async_copy_b128(lds_nxt + 16, gsrc + 8);
            gsrc += (size_t)32 * ldb;
        }
#pragma unroll
        for (int j = 0; j < 4; ++j) {
            v16h b;
#pragma unroll
            for (int i = 0; i < 16; ++i)
                b[i] = bb_cur[((i & 7) + 8 * (lane >> 4) + ((i >> 3) << 4)) * 64 + j * 16];
            acc[0][j] = wmma16(a0, b, acc[0][j]);
            acc[1][j] = wmma16(a1, b, acc[1][j]);
        }
        uint32_t tl = lds_cur; lds_cur = lds_nxt; lds_nxt = tl;
        const _Float16* tp = bb_cur; bb_cur = bb_nxt; bb_nxt = tp;
    }

    int lh = lane >> 4, nl = lane & 15;
#pragma unroll
    for (int mi = 0; mi < 2; ++mi) {
#pragma unroll
        for (int j = 0; j < 4; ++j) {
            int n = n0 + j * 16 + nl;
            float bv = bias[n];
#pragma unroll
            for (int e = 0; e < 8; ++e) {
                int t = row0 + mi * 16 + e + 8 * lh;   // global token
                int bi = t >> 11, tt = t & (SEQ - 1);
                _Float16 h = (_Float16)(acc[mi][j][e] + bv);
                if (n < N_EMBD) {
                    int hd = n >> 6, d = n & 63;
                    Q[(((size_t)bi * N_HEADS + hd) * SEQ + tt) * HEAD + d] = h;
                } else if (n < 2 * N_EMBD) {
                    int c = n - N_EMBD, hd = c >> 6, d = c & 63;
                    Kt[(((size_t)bi * N_HEADS + hd) * HEAD + d) * SEQ + tt] = h;
                } else {
                    int c = n - 2 * N_EMBD, hd = c >> 6, d = c & 63;
                    V[(((size_t)bi * N_HEADS + hd) * SEQ + tt) * HEAD + d] = h;
                }
            }
        }
    }
}

// Flash attention: one wave handles one (b,h) x 16-query tile.
__global__ __launch_bounds__(128)
void attn_kernel(const _Float16* __restrict__ Q, const _Float16* __restrict__ Kt,
                 const _Float16* __restrict__ V, _Float16* __restrict__ Obuf) {
    __shared__ __align__(64) _Float16 pbuf[4][16 * 32];
    int wave = threadIdx.x >> 5;
    int lane = threadIdx.x & 31;
    int task = blockIdx.x * 4 + wave;        // 0 .. 8191
    int q0   = (task & 127) * 16;
    int bh   = task >> 7;                    // 0 .. 63
    const _Float16* Qp = Q  + (size_t)bh * SEQ * HEAD;
    const _Float16* Kp = Kt + (size_t)bh * HEAD * SEQ;
    const _Float16* Vp = V  + (size_t)bh * SEQ * HEAD;
    _Float16* pl = pbuf[wave];

    v16h qa0 = load_a(Qp, HEAD, q0, 0);
    v16h qa1 = load_a(Qp, HEAD, q0, 32);

    v8f o[4] = {};
    float mrow[8], lrow[8];
#pragma unroll
    for (int e = 0; e < 8; ++e) { mrow[e] = -3.0e38f; lrow[e] = 0.f; }

    int lh = lane >> 4, nl = lane & 15;
    int nb = (q0 + 47) >> 5;                 // key blocks of 32 covering 0..q0+15
    for (int kb = 0; kb < nb; ++kb) {
        int kk0 = kb << 5;
        v8f s0 = {}, s1 = {};
        s0 = wmma16(qa0, load_b(Kp, SEQ, 0,  kk0),      s0);
        s0 = wmma16(qa1, load_b(Kp, SEQ, 32, kk0),      s0);
        s1 = wmma16(qa0, load_b(Kp, SEQ, 0,  kk0 + 16), s1);
        s1 = wmma16(qa1, load_b(Kp, SEQ, 32, kk0 + 16), s1);
        bool edge = (kk0 + 31 > q0);
#pragma unroll
        for (int e = 0; e < 8; ++e) {
            int qrow = q0 + e + 8 * lh;
            float v0 = s0[e] * 0.125f;       // 1/sqrt(64)
            float v1 = s1[e] * 0.125f;
            if (edge) {
                if (kk0 + nl > qrow)      v0 = -1.0e9f;
                if (kk0 + 16 + nl > qrow) v1 = -1.0e9f;
            }
            float rm = fmaxf(v0, v1);
#pragma unroll
            for (int msk = 1; msk < 16; msk <<= 1)
                rm = fmaxf(rm, __shfl_xor(rm, msk, 32));
            float mnew = fmaxf(mrow[e], rm);
            float corr = __expf(mrow[e] - mnew);
            mrow[e] = mnew;
            float p0 = __expf(v0 - mnew);
            float p1 = __expf(v1 - mnew);
            float rs = p0 + p1;
#pragma unroll
            for (int msk = 1; msk < 16; msk <<= 1)
                rs += __shfl_xor(rs, msk, 32);
            lrow[e] = lrow[e] * corr + rs;
#pragma unroll
            for (int j = 0; j < 4; ++j) o[j][e] *= corr;
            int m = e + 8 * lh;
            pl[m * 32 + nl]      = (_Float16)p0;
            pl[m * 32 + 16 + nl] = (_Float16)p1;
        }
        // Re-layout P (D-fragment in LDS, row-major 16x32) -> A-fragment.
        // Same-wave DS ops are in-order; per-wave buffer, no barrier needed.
        {
            int m = lane & 15;
            const _Float16* pp = pl + m * 32 + 8 * lh;
            v8h lo = *(const v8h*)pp;
            v8h hi = *(const v8h*)(pp + 16);
            v16h pa;
#pragma unroll
            for (int i = 0; i < 8; ++i) { pa[i] = lo[i]; pa[8 + i] = hi[i]; }
#pragma unroll
            for (int j = 0; j < 4; ++j)
                o[j] = wmma16(pa, load_b(Vp, HEAD, kk0, j * 16), o[j]);
        }
    }
    // Normalize and store to attn buffer [TOKENS x N_EMBD] (f16), heads interleaved.
    int b = bh >> 4, h = bh & 15;
#pragma unroll
    for (int j = 0; j < 4; ++j) {
        int col = h * HEAD + j * 16 + nl;
#pragma unroll
        for (int e = 0; e < 8; ++e) {
            int t = b * SEQ + q0 + e + 8 * lh;
            Obuf[(size_t)t * N_EMBD + col] = (_Float16)(o[j][e] / lrow[e]);
        }
    }
}

// out = attn @ W_out + b_out  (f32 output)
__global__ __launch_bounds__(128)
void out_gemm(const _Float16* __restrict__ A, const _Float16* __restrict__ W,
              const float* __restrict__ bias, float* __restrict__ out) {
    __shared__ __align__(128) _Float16 sB[2][32 * 64];
    const int ldb = N_EMBD;
    const int NK = N_EMBD / 32;
    int tid  = threadIdx.x;
    int wave = tid >> 5;
    int lane = tid & 31;
    int row0 = blockIdx.y * 128 + wave * 32;
    int n0   = blockIdx.x * 64;

    int byt  = tid * 32;
    int brow = byt >> 7;
    int bcol = (byt & 127) >> 1;

    uint32_t lds_cur = (uint32_t)(size_t)&sB[0][0] + byt;
    uint32_t lds_nxt = (uint32_t)(size_t)&sB[1][0] + byt;
    const _Float16* bb_cur = &sB[0][0] + (lane & 15);
    const _Float16* bb_nxt = &sB[1][0] + (lane & 15);
    const _Float16* gsrc = W + (size_t)brow * ldb + n0 + bcol;

    async_copy_b128(lds_cur, gsrc);
    async_copy_b128(lds_cur + 16, gsrc + 8);
    gsrc += (size_t)32 * ldb;

    v8f acc[2][4] = {};
#pragma clang loop unroll(disable)
    for (int kb = 0; kb < NK; ++kb) {
        v16h a0 = load_a(A, N_EMBD, row0,      kb * 32);
        v16h a1 = load_a(A, N_EMBD, row0 + 16, kb * 32);
        wait_async0();
        __syncthreads();
        if (kb + 1 < NK) {
            async_copy_b128(lds_nxt, gsrc);
            async_copy_b128(lds_nxt + 16, gsrc + 8);
            gsrc += (size_t)32 * ldb;
        }
#pragma unroll
        for (int j = 0; j < 4; ++j) {
            v16h b;
#pragma unroll
            for (int i = 0; i < 16; ++i)
                b[i] = bb_cur[((i & 7) + 8 * (lane >> 4) + ((i >> 3) << 4)) * 64 + j * 16];
            acc[0][j] = wmma16(a0, b, acc[0][j]);
            acc[1][j] = wmma16(a1, b, acc[1][j]);
        }
        uint32_t tl = lds_cur; lds_cur = lds_nxt; lds_nxt = tl;
        const _Float16* tp = bb_cur; bb_cur = bb_nxt; bb_nxt = tp;
    }

    int lh = lane >> 4, nl = lane & 15;
#pragma unroll
    for (int mi = 0; mi < 2; ++mi) {
#pragma unroll
        for (int j = 0; j < 4; ++j) {
            int n = n0 + j * 16 + nl;
            float bv = bias[n];
#pragma unroll
            for (int e = 0; e < 8; ++e) {
                int t = row0 + mi * 16 + e + 8 * lh;
                out[(size_t)t * N_EMBD + n] = acc[mi][j][e] + bv;
            }
        }
    }
}

extern "C" void kernel_launch(void* const* d_in, const int* in_sizes, int n_in,
                              void* d_out, int out_size, void* d_ws, size_t ws_size,
                              hipStream_t stream) {
    (void)in_sizes; (void)n_in; (void)out_size; (void)ws_size;
    const float* x    = (const float*)d_in[0];
    const float* Wqkv = (const float*)d_in[1];
    const float* bqkv = (const float*)d_in[2];
    const float* Wout = (const float*)d_in[3];
    const float* bout = (const float*)d_in[4];

    // f16 workspace layout (element offsets)
    _Float16* xh  = (_Float16*)d_ws;                               // 8192*1024
    _Float16* wqh = xh  + (size_t)TOKENS * N_EMBD;                 // 1024*3072
    _Float16* woh = wqh + (size_t)N_EMBD * 3 * N_EMBD;             // 1024*1024
    _Float16* Q   = woh + (size_t)N_EMBD * N_EMBD;                 // 8388608
    _Float16* Kt  = Q   + (size_t)TOKENS * N_EMBD;
    _Float16* V   = Kt  + (size_t)TOKENS * N_EMBD;
    _Float16* att = V   + (size_t)TOKENS * N_EMBD;

    cvt_f32_f16<<<4096, 256, 0, stream>>>(x, xh, TOKENS * N_EMBD);
    cvt_f32_f16<<<2048, 256, 0, stream>>>(Wqkv, wqh, N_EMBD * 3 * N_EMBD);
    cvt_f32_f16<<<1024, 256, 0, stream>>>(Wout, woh, N_EMBD * N_EMBD);

    dim3 g1(3 * N_EMBD / 64, TOKENS / 128);  // 48 x 64
    qkv_gemm<<<g1, 128, 0, stream>>>(xh, wqh, bqkv, Q, Kt, V);

    attn_kernel<<<(BATCH * N_HEADS * (SEQ / 16)) / 4, 128, 0, stream>>>(Q, Kt, V, att);

    dim3 g2(N_EMBD / 64, TOKENS / 128);      // 16 x 64
    out_gemm<<<g2, 128, 0, stream>>>(att, woh, bout, (float*)d_out);
}
